// VideoAttention_48782238548619
// MI455X (gfx1250) — matrix operand order repaired and checked by reading, not compiled, not run on hardware
//
#include <hip/hip_runtime.h>
#include <hip/hip_bf16.h>
#include <math.h>
#include <stdint.h>

// ---- problem constants (batch_size=2 hardcoded per reference setup) ----
#define C_IN     512
#define O_QKV    1536
#define N_TOK    4096      // 16 images * 256 pixels
#define S_LEN    2048      // 8 frames * 256 pixels per (batch, head)
#define HEAD     64
#define N_BM     16        // batch(2) * heads(8)

typedef _Float16 half8 __attribute__((ext_vector_type(8)));
typedef _Float16 v16h  __attribute__((ext_vector_type(16)));
typedef float    v8f   __attribute__((ext_vector_type(8)));

// ---------------- WMMA helpers (CDNA5 16x16x32 f16 -> f32) ----------------
static __device__ __forceinline__ v8f wmma_f16(v16h a, v16h b, v8f c) {
  // D = A(16x32) * B(32x16) + C ; args: (neg_a, A, neg_b, B, c_mod, C, reuse_a, reuse_b)
  return __builtin_amdgcn_wmma_f32_16x16x32_f16(false, a, false, b, (short)0, c, false, false);
}

static __device__ __forceinline__ v16h combine16(half8 lo, half8 hi) {
  v16h r;
#pragma unroll
  for (int i = 0; i < 8; ++i) { r[i] = lo[i]; r[i + 8] = hi[i]; }
  return r;
}

// A-matrix fragment (16xK-chunk of 32). base = row start + k0 for this lane's row.
// Lane L: M = L%16 ; halves i -> K = i + 16*(L>=16)
static __device__ __forceinline__ v16h ld_a_frag(const _Float16* base, int L) {
  int b = (L & 16) ? 16 : 0;
  half8 lo = *(const half8*)(base + b);
  half8 hi = *(const half8*)(base + 8 + b);
  return combine16(lo, hi);
}

// B-matrix fragment from 16 contiguous halves: lane L: N = L%16 ; halves i -> K = i + 16*(L>=16)
// Caller points p at (row-for-this-lane) + 16*(L>=16).
static __device__ __forceinline__ v16h ld_b_contig(const _Float16* p) {
  half8 lo = *(const half8*)p;
  half8 hi = *(const half8*)(p + 8);
  return combine16(lo, hi);
}

// ---------------- CDNA5 async copy: global -> LDS, 16B per lane ----------------
// GLOBAL_LOAD_ASYNC_TO_LDS_B128 (VGLOBAL op 98), tracked with ASYNCcnt.
// LDS destination is the allocation-relative byte offset (low 32 bits of the
// generic pointer to a __shared__ object); HW adds the wave's LDS_BASE.
static __device__ __forceinline__ void async_ld_lds_b128(void* lptr, const void* gptr) {
  uint32_t loff = (uint32_t)(uintptr_t)lptr;
  uint64_t gaddr = (uint64_t)(uintptr_t)gptr;
  asm volatile("global_load_async_to_lds_b128 %0, %1, off"
               :: "v"(loff), "v"(gaddr)
               : "memory");
}
static __device__ __forceinline__ void wait_async0() {
  asm volatile("s_wait_asynccnt 0" ::: "memory");
}

// ---------------- K1: magnitude-preserving weight normalization -> f16 ----------------
__global__ void wnorm_kernel(const float* __restrict__ w_qkv, const float* __restrict__ w_proj,
                             _Float16* __restrict__ wq_h, _Float16* __restrict__ wp_h) {
  int row = blockIdx.x * 8 + (threadIdx.x >> 5);
  int L = threadIdx.x & 31;
  const float* src; _Float16* dst;
  if (row < O_QKV) { src = w_qkv + (size_t)row * C_IN;            dst = wq_h + (size_t)row * C_IN; }
  else             { src = w_proj + (size_t)(row - O_QKV) * C_IN; dst = wp_h + (size_t)(row - O_QKV) * C_IN; }
  float ss = 0.f;
#pragma unroll
  for (int i = L; i < C_IN; i += 32) { float v = src[i]; ss += v * v; }
#pragma unroll
  for (int m = 16; m >= 1; m >>= 1) ss += __shfl_xor(ss, m, 32);
  const float inv_sqrt_fan = 0.044194173824159216f; // 1/sqrt(512)
  float norm = sqrtf(ss);
  float scale = inv_sqrt_fan / (1e-4f + norm * inv_sqrt_fan); // w/(eps+|w|/sqrt(f))/sqrt(f)
#pragma unroll
  for (int i = L; i < C_IN; i += 32) dst[i] = (_Float16)(src[i] * scale);
}

// ---------------- K2: x (n,C,h,w) -> token-major f16 [tok][C] ----------------
__global__ void xcast_kernel(const float* __restrict__ x, _Float16* __restrict__ x_h) {
  int gid = blockIdx.x * 256 + threadIdx.x;     // N_TOK * C_IN = 2M
  int cc = gid & (C_IN - 1);
  int tok = gid >> 9;
  int n = tok >> 8, p = tok & 255;
  x_h[(size_t)tok * C_IN + cc] = (_Float16)x[((size_t)n * C_IN + cc) * 256 + p];
}

// ---------------- K3: QKV GEMM  qkv[tok][o] = x_h @ w_qkv^T ----------------
// wave: 16 tokens x 64 outputs; K = 512 in chunks of 32; software-pipelined.
__global__ void gemm_qkv_kernel(const _Float16* __restrict__ x_h, const _Float16* __restrict__ w_h,
                                _Float16* __restrict__ qkv_h) {
  int wv = blockIdx.x * 8 + (threadIdx.x >> 5);
  int L = threadIdx.x & 31;
  int tokTile = wv / 24;              // 256 token tiles
  int oq = wv % 24;                   // 24 quads of 4 o-tiles
  int t0 = tokTile * 16, o0 = oq * 64;
  const _Float16* arow = x_h + (size_t)(t0 + (L & 15)) * C_IN;
  const _Float16* b0 = w_h + (size_t)(o0 + (L & 15)) * C_IN + ((L & 16) ? 16 : 0);
  v8f a0 = {}, a1 = {}, a2 = {}, a3 = {};
  // prologue: fragments for k0 = 0
  v16h a  = ld_a_frag(arow, L);
  v16h f0 = ld_b_contig(b0);
  v16h f1 = ld_b_contig(b0 + 16 * C_IN);
  v16h f2 = ld_b_contig(b0 + 32 * C_IN);
  v16h f3 = ld_b_contig(b0 + 48 * C_IN);
#pragma unroll 5
  for (int k0 = 32; k0 < C_IN; k0 += 32) {
    // issue next chunk's loads before consuming current fragments
    v16h an = ld_a_frag(arow + k0, L);
    v16h g0 = ld_b_contig(b0 + k0);
    v16h g1 = ld_b_contig(b0 + 16 * C_IN + k0);
    v16h g2 = ld_b_contig(b0 + 32 * C_IN + k0);
    v16h g3 = ld_b_contig(b0 + 48 * C_IN + k0);
    a0 = wmma_f16(a, f0, a0);
    a1 = wmma_f16(a, f1, a1);
    a2 = wmma_f16(a, f2, a2);
    a3 = wmma_f16(a, f3, a3);
    a = an; f0 = g0; f1 = g1; f2 = g2; f3 = g3;
  }
  a0 = wmma_f16(a, f0, a0);
  a1 = wmma_f16(a, f1, a1);
  a2 = wmma_f16(a, f2, a2);
  a3 = wmma_f16(a, f3, a3);
#pragma unroll
  for (int r = 0; r < 8; ++r) {
    int tok = t0 + r + ((L & 16) ? 8 : 0);
    _Float16* out = qkv_h + (size_t)tok * O_QKV + o0 + (L & 15);
    out[0]  = (_Float16)a0[r];
    out[16] = (_Float16)a1[r];
    out[32] = (_Float16)a2[r];
    out[48] = (_Float16)a3[r];
  }
}

// ---------------- K4: RoPE + scatter to [bm][s][c] f16 q/k/v ----------------
__global__ void rope_kernel(const _Float16* __restrict__ qkv, _Float16* __restrict__ qh,
                            _Float16* __restrict__ kh, _Float16* __restrict__ vh) {
  int gid = blockIdx.x * 256 + threadIdx.x;  // 4096 tok * 8 heads * 32 pairs = 1M
  int j = gid & 31;
  int m = (gid >> 5) & 7;
  int tok = gid >> 8;
  int n = tok >> 8, p = tok & 255;
  int b = n >> 3, t = n & 7;
  int s = t * 256 + p;
  int bm = b * 8 + m;
  const _Float16* row = qkv + (size_t)tok * O_QKV + m * HEAD;
  float q1 = (float)row[j],        q2 = (float)row[j + 32];
  float k1 = (float)row[512 + j],  k2 = (float)row[512 + j + 32];
  float v1 = (float)row[1024 + j], v2 = (float)row[1024 + j + 32];
  float inv_freq = __powf(10000.f, -(float)j * (1.0f / 32.0f));
  float ang = (float)s * inv_freq;
  float cs = __cosf(ang), sn = __sinf(ang);
  size_t o = ((size_t)bm * S_LEN + s) * HEAD + j;
  qh[o] = (_Float16)(q1 * cs - q2 * sn); qh[o + 32] = (_Float16)(q1 * sn + q2 * cs);
  kh[o] = (_Float16)(k1 * cs - k2 * sn); kh[o + 32] = (_Float16)(k1 * sn + k2 * cs);
  vh[o] = (_Float16)v1;                  vh[o + 32] = (_Float16)v2;
}

// ---------------- K5: flash attention, frame-causal ----------------
// block = 8 waves = 128 q rows (one frame slice). Each wave owns a 16-row q tile.
__global__ void attn_kernel(const _Float16* __restrict__ qh, const _Float16* __restrict__ kh,
                            const _Float16* __restrict__ vh, _Float16* __restrict__ attn_h) {
  __shared__ _Float16 lds_k[32 * 64];     // K chunk, row-major [k][c]
  __shared__ _Float16 lds_vT[64 * 32];    // V chunk transposed [c][k]
  __shared__ _Float16 lds_p[8][16 * 32];  // per-wave P tile scratch
  int blk = blockIdx.x;
  int bm = blk >> 4;                       // 0..15
  int qg = blk & 15;                       // 128-row q group
  int tid = threadIdx.x;
  int wave = tid >> 5;
  int L = tid & 31;
  const _Float16* qb = qh + (size_t)bm * S_LEN * HEAD;
  const _Float16* kb = kh + (size_t)bm * S_LEN * HEAD;
  const _Float16* vb = vh + (size_t)bm * S_LEN * HEAD;
  int s0 = qg * 128 + wave * 16;
  int frame = qg >> 1;                     // 256-row frames, 128-row groups
  int kEnd = (frame + 1) << 8;             // visible keys: [0, (frame+1)*256)

  const _Float16* qrow = qb + (size_t)(s0 + (L & 15)) * HEAD;
  v16h qA0 = ld_a_frag(qrow, L);           // c 0..31
  v16h qA1 = ld_a_frag(qrow + 32, L);      // c 32..63

  v8f o0 = {}, o1 = {}, o2 = {}, o3 = {};
  float mrow[8], lrow[8];
#pragma unroll
  for (int r = 0; r < 8; ++r) { mrow[r] = -1e30f; lrow[r] = 0.f; }
  const float scale = 0.125f;              // 1/sqrt(64)

  int srow = tid >> 3;                     // 32 rows, 8 threads/row
  int scol = (tid & 7) * 8;                // 8 halves (16B) per thread

  for (int k0 = 0; k0 < kEnd; k0 += 32) {
    __syncthreads();                       // previous chunk fully consumed
    // K chunk: async DMA global -> LDS (16B per lane, ASYNCcnt-tracked)
    async_ld_lds_b128(&lds_k[srow * 64 + scol],
                      kb + (size_t)(k0 + srow) * HEAD + scol);
    // V chunk: through VGPRs because it is stored transposed
    {
      half8 vv = *(const half8*)(vb + (size_t)(k0 + srow) * HEAD + scol);
#pragma unroll
      for (int e = 0; e < 8; ++e) lds_vT[(scol + e) * 32 + srow] = vv[e];
    }
    if (k0 + 32 < kEnd) __builtin_prefetch(kb + (size_t)(k0 + 32) * HEAD, 0, 0);
    wait_async0();                         // our async writes landed in LDS
    __syncthreads();                       // everyone's writes visible

    // scores: two 16x16 tiles covering k cols [k0, k0+32)
    int bsel = (L & 16) ? 16 : 0;
    const _Float16* kr0 = &lds_k[(L & 15) * 64 + bsel];
    const _Float16* kr1 = &lds_k[((L & 15) + 16) * 64 + bsel];
    v8f s0v = {}, s1v = {};
    s0v = wmma_f16(qA0, ld_b_contig(kr0), s0v);
    s0v = wmma_f16(qA1, ld_b_contig(kr0 + 32), s0v);
    s1v = wmma_f16(qA0, ld_b_contig(kr1), s1v);
    s1v = wmma_f16(qA1, ld_b_contig(kr1 + 32), s1v);

    // online softmax over this 32-col slab (rows split across 16-lane halves)
#pragma unroll
    for (int r = 0; r < 8; ++r) {
      float a = s0v[r] * scale, b = s1v[r] * scale;
      float rm = fmaxf(a, b);
#pragma unroll
      for (int msk = 1; msk < 16; msk <<= 1) rm = fmaxf(rm, __shfl_xor(rm, msk, 32));
      float mnew = fmaxf(mrow[r], rm);
      float corr = __expf(mrow[r] - mnew);
      float p0 = __expf(a - mnew);
      float p1 = __expf(b - mnew);
      float rs = p0 + p1;
#pragma unroll
      for (int msk = 1; msk < 16; msk <<= 1) rs += __shfl_xor(rs, msk, 32);
      lrow[r] = lrow[r] * corr + rs;
      mrow[r] = mnew;
      o0[r] *= corr; o1[r] *= corr; o2[r] *= corr; o3[r] *= corr;
      int prow = r + ((L & 16) ? 8 : 0);
      lds_p[wave][prow * 32 + (L & 15)] = (_Float16)p0;
      lds_p[wave][prow * 32 + 16 + (L & 15)] = (_Float16)p1;
    }
    // D-layout -> A-layout via per-wave LDS round trip
    v16h pA = ld_a_frag(&lds_p[wave][(L & 15) * 32], L);
    o0 = wmma_f16(pA, ld_b_contig(&lds_vT[( 0 + (L & 15)) * 32 + bsel]), o0);
    o1 = wmma_f16(pA, ld_b_contig(&lds_vT[(16 + (L & 15)) * 32 + bsel]), o1);
    o2 = wmma_f16(pA, ld_b_contig(&lds_vT[(32 + (L & 15)) * 32 + bsel]), o2);
    o3 = wmma_f16(pA, ld_b_contig(&lds_vT[(48 + (L & 15)) * 32 + bsel]), o3);
  }

  // normalize + scatter back to token-major [tok][512] f16 for projection GEMM
  int b = bm >> 3, m = bm & 7;
#pragma unroll
  for (int r = 0; r < 8; ++r) {
    int sr = s0 + r + ((L & 16) ? 8 : 0);
    int t = sr >> 8, p = sr & 255;
    int tok = (b * 8 + t) * 256 + p;
    float inv = 1.0f / lrow[r];
    _Float16* out = attn_h + (size_t)tok * C_IN + m * HEAD + (L & 15);
    out[0]  = (_Float16)(o0[r] * inv);
    out[16] = (_Float16)(o1[r] * inv);
    out[32] = (_Float16)(o2[r] * inv);
    out[48] = (_Float16)(o3[r] * inv);
  }
}

// ---------------- K6: projection GEMM + residual epilogue (fp32 NCHW out) ----------------
__global__ void gemm_proj_kernel(const _Float16* __restrict__ attn_h, const _Float16* __restrict__ w_h,
                                 const float* __restrict__ x, float* __restrict__ out) {
  int wv = blockIdx.x * 8 + (threadIdx.x >> 5);
  int L = threadIdx.x & 31;
  int tokTile = wv >> 3;               // 256 token tiles
  int oq = wv & 7;                     // 8 quads of 4 o-tiles (512 outputs)
  int t0 = tokTile * 16, o0 = oq * 64;
  const _Float16* arow = attn_h + (size_t)(t0 + (L & 15)) * C_IN;
  const _Float16* b0 = w_h + (size_t)(o0 + (L & 15)) * C_IN + ((L & 16) ? 16 : 0);
  v8f acc[4] = {{}, {}, {}, {}};
  v16h a  = ld_a_frag(arow, L);
  v16h f0 = ld_b_contig(b0);
  v16h f1 = ld_b_contig(b0 + 16 * C_IN);
  v16h f2 = ld_b_contig(b0 + 32 * C_IN);
  v16h f3 = ld_b_contig(b0 + 48 * C_IN);
#pragma unroll 5
  for (int k0 = 32; k0 < C_IN; k0 += 32) {
    v16h an = ld_a_frag(arow + k0, L);
    v16h g0 = ld_b_contig(b0 + k0);
    v16h g1 = ld_b_contig(b0 + 16 * C_IN + k0);
    v16h g2 = ld_b_contig(b0 + 32 * C_IN + k0);
    v16h g3 = ld_b_contig(b0 + 48 * C_IN + k0);
    acc[0] = wmma_f16(a, f0, acc[0]);
    acc[1] = wmma_f16(a, f1, acc[1]);
    acc[2] = wmma_f16(a, f2, acc[2]);
    acc[3] = wmma_f16(a, f3, acc[3]);
    a = an; f0 = g0; f1 = g1; f2 = g2; f3 = g3;
  }
  acc[0] = wmma_f16(a, f0, acc[0]);
  acc[1] = wmma_f16(a, f1, acc[1]);
  acc[2] = wmma_f16(a, f2, acc[2]);
  acc[3] = wmma_f16(a, f3, acc[3]);
  const float inv_mix = 1.3130643285972254f; // 1/sqrt(0.7^2 + 0.3^2)
#pragma unroll
  for (int r = 0; r < 8; ++r) {
    int tok = t0 + r + ((L & 16) ? 8 : 0);
    int n = tok >> 8, p = tok & 255;
#pragma unroll
    for (int nt = 0; nt < 4; ++nt) {
      int ch = o0 + nt * 16 + (L & 15);
      size_t xi = ((size_t)n * C_IN + ch) * 256 + p;
      out[xi] = (0.7f * x[xi] + 0.3f * acc[nt][r]) * inv_mix;
    }
  }
}

// ---------------- launch ----------------
extern "C" void kernel_launch(void* const* d_in, const int* in_sizes, int n_in,
                              void* d_out, int out_size, void* d_ws, size_t ws_size,
                              hipStream_t stream) {
  const float* x      = (const float*)d_in[0];
  const float* w_qkv  = (const float*)d_in[1];
  const float* w_proj = (const float*)d_in[2];
  float* out = (float*)d_out;
  char* ws = (char*)d_ws;
  // workspace layout (all offsets 256B-aligned), total ~34 MB
  _Float16* wq_h   = (_Float16*)(ws + 0);          // 1536*512 f16
  _Float16* wp_h   = (_Float16*)(ws + 1572864);    // 512*512  f16
  _Float16* x_h    = (_Float16*)(ws + 2097152);    // 4096*512 f16
  _Float16* qkv_h  = (_Float16*)(ws + 6291456);    // 4096*1536 f16
  _Float16* q_h    = (_Float16*)(ws + 18874368);   // 16*2048*64 f16
  _Float16* k_h    = (_Float16*)(ws + 23068672);
  _Float16* v_h    = (_Float16*)(ws + 27262976);
  _Float16* attn_h = (_Float16*)(ws + 31457280);   // 4096*512 f16

  wnorm_kernel    <<<256, 256, 0, stream>>>(w_qkv, w_proj, wq_h, wp_h);
  xcast_kernel    <<<8192, 256, 0, stream>>>(x, x_h);
  gemm_qkv_kernel <<<768, 256, 0, stream>>>(x_h, wq_h, qkv_h);
  rope_kernel     <<<4096, 256, 0, stream>>>(qkv_h, q_h, k_h, v_h);
  attn_kernel     <<<256, 256, 0, stream>>>(q_h, k_h, v_h, attn_h);
  gemm_proj_kernel<<<256, 256, 0, stream>>>(attn_h, wp_h, x, out);
}
